// DerivNet2D_v2_63591285784703
// MI455X (gfx1250) — matrix operand order repaired and checked by reading, not compiled, not run on hardware
//
#include <hip/hip_runtime.h>

// DerivNet2D fused kernel for gfx1250 (MI455X).
// Three M=32768,N=1024,K=1024 GEMMs sharing one B (w2.T); A tiles generated on
// the fly from layer-1; epilogue fully fused -> 3 floats out per row.
// B slabs stream into LDS with global_load_async_to_lds_b128 (double-buffered)
// from an f16 copy of w2 made once per launch in d_ws.

#define NXC   32768
#define NH    1024
#define MTILE 32          // rows per workgroup
#define NTILE 128         // h2-features per n-block (4 wave-cols x 32)
#define KSLAB 64          // K staged per barrier period
#define ASTRIDE 1032      // halves per A row (1024 + pad, keeps 16B align)
#define BSTRIDE 72        // halves per B column (64 + pad, keeps 16B align)
#define NSLAB (NH / KSLAB)

typedef _Float16 v8h  __attribute__((ext_vector_type(8)));
typedef _Float16 v16h __attribute__((ext_vector_type(16)));
typedef float    v8f  __attribute__((ext_vector_type(8)));

__device__ __forceinline__ v16h frag_cat(v8h lo, v8h hi) {
  return __builtin_shufflevector(lo, hi, 0,1,2,3,4,5,6,7,8,9,10,11,12,13,14,15);
}

// A 16x32 f16 fragment (ISA 7.12.2): lanes 0-15 -> K[0:8)+K[16:24),
// lanes 16-31 -> K[8:16)+K[24:32); row = rLoc + (lane&15).
__device__ __forceinline__ v16h load_a(const _Float16* As, int rLoc, int k0, int lane) {
  const _Float16* p = As + (rLoc + (lane & 15)) * ASTRIDE + k0 + ((lane >> 4) << 3);
  v8h lo = *(const v8h*)(p);
  v8h hi = *(const v8h*)(p + 16);
  return frag_cat(lo, hi);
}

// B 32x16 f16 fragment: col n = lane&15 (+base); lanes 0-15 hold K[0:8)+K[16:24),
// lanes 16-31 hold K[8:16)+K[24:32) (same order as A).
__device__ __forceinline__ v16h load_b(const _Float16* Bl, int nBase, int kk, int lane) {
  const _Float16* p = Bl + (nBase + (lane & 15)) * BSTRIDE + kk + ((lane >> 4) << 3);
  v8h lo = *(const v8h*)(p);
  v8h hi = *(const v8h*)(p + 16);
  return frag_cat(lo, hi);
}

// ---- gfx1250 async copy: global -> LDS, 16 bytes, tracked by ASYNCcnt ------
__device__ __forceinline__ void async_ld_b128(unsigned lds_off, const void* g) {
  asm volatile("global_load_async_to_lds_b128 %0, %1, off"
               :: "v"(lds_off), "v"(g) : "memory");
}
__device__ __forceinline__ void wait_async0() {
  asm volatile("s_wait_asynccnt 0x0" ::: "memory");
}
// Generic pointer into LDS -> LDS byte offset (ISA 10.2: aperture keeps
// the offset in addr[31:0]).
__device__ __forceinline__ unsigned lds_off(const void* p) {
  return (unsigned)(uintptr_t)p;
}

__device__ __forceinline__ v8f wmma_f16(v16h a, v16h b, v8f c) {
  return __builtin_amdgcn_wmma_f32_16x16x32_f16(false, a, false, b,
                                                (short)0, c, false, false);
}

// ---- one-shot w2 f32 -> f16 conversion into workspace ----------------------
__global__ void __launch_bounds__(256)
convert_w2_f16(const float* __restrict__ w2, _Float16* __restrict__ w2h) {
  int idx = (blockIdx.x * 256 + threadIdx.x) * 8;
  const float4* s = (const float4*)(w2 + idx);
  float4 a = s[0], b = s[1];
  v8h h = { (_Float16)a.x, (_Float16)a.y, (_Float16)a.z, (_Float16)a.w,
            (_Float16)b.x, (_Float16)b.y, (_Float16)b.z, (_Float16)b.w };
  *(v8h*)(w2h + idx) = h;
}

__global__ void __launch_bounds__(256)
derivnet2d_fused(const float* __restrict__ x,   const float* __restrict__ w1,
                 const float* __restrict__ w1_2, const float* __restrict__ b1,
                 const float* __restrict__ w2,  const float* __restrict__ b2,
                 const float* __restrict__ w3,  const float* __restrict__ b3,
                 const _Float16* __restrict__ w2h, int useAsync,
                 float* __restrict__ out)
{
  extern __shared__ char smem[];
  _Float16* As0 = (_Float16*)smem;                 // z1^2           (GEMM-1 A)
  _Float16* As1 = As0 + MTILE * ASTRIDE;           // z1*(1-z1^2)*dh1dx1
  _Float16* As2 = As1 + MTILE * ASTRIDE;           // z1*(1-z1^2)*dh1dx2
  _Float16* Bl  = As2 + MTILE * ASTRIDE;           // double-buffered w2 slabs
  float* xs    = (float*)(Bl + 2 * NTILE * BSTRIDE);
  float* yAcc  = xs + 2 * MTILE;
  float* dAcc1 = yAcc + MTILE;
  float* dAcc2 = dAcc1 + MTILE;

  const int t    = threadIdx.x;
  const int lane = t & 31;
  const int wv   = t >> 5;             // 8 waves: 2 (M) x 4 (N)
  const int rLoc = (wv & 1) * 16;      // wave row base within 32-row tile
  const int nWc  = (wv >> 1) * 32;     // wave col base within 128-col block
  const int j0   = blockIdx.x * MTILE;

  if (t < 2 * MTILE) xs[t] = x[j0 * 2 + t];
  if (t < MTILE) { yAcc[t] = 0.f; dAcc1[t] = 0.f; dAcc2[t] = 0.f; }
  __syncthreads();

  // ---- Phase A: build the three A tiles (layer-1 fused, f16) --------------
  for (int e = t; e < MTILE * NH; e += 256) {
    int i = e & (NH - 1), j = e >> 10;
    float x0 = xs[2*j], x1 = xs[2*j + 1];
    float wa = w1[2*i],   wb = w1[2*i + 1];
    float va = w1_2[2*i], vb = w1_2[2*i + 1];
    float h1 = fmaf(x0*x0, wa, fmaf(x1*x1, wb, fmaf(x0, va, fmaf(x1, vb, b1[i]))));
    float z1 = tanhf(h1);
    float dd = z1 * (1.f - z1*z1);                 // z1 * dz1dh1
    As0[j*ASTRIDE + i] = (_Float16)(z1 * z1);
    As1[j*ASTRIDE + i] = (_Float16)(dd * fmaf(wa, x0, va));   // dh1dx1 branch
    As2[j*ASTRIDE + i] = (_Float16)(dd * fmaf(wb, x1, vb));   // dh1dx2 branch
  }
  __syncthreads();

  const int sCol = t & (NTILE - 1);    // staging column within n-block
  const int sK   = (t >> 7) * 32;      // staging k base: 0 or 32 (32 halves/thread)

  // ---- Phase B: n-block sweep, shared-B 6-WMMA steps + fused epilogue -----
  for (int nb = 0; nb < NH / NTILE; ++nb) {
    v8f acc0[2] = {{}, {}}, acc1[2] = {{}, {}}, acc2[2] = {{}, {}};
    const int mBase = nb * NTILE;
    const size_t srow = (size_t)(mBase + sCol) * NH;

    // stage slab s into buffer (s&1)
    auto stage = [&](int s) {
      _Float16* dst = Bl + (s & 1) * NTILE * BSTRIDE + sCol * BSTRIDE + sK;
      int ks = s * KSLAB;
      if (useAsync) {
        const _Float16* g = w2h + srow + ks + sK;
        unsigned l = lds_off(dst);
        async_ld_b128(l,      g);
        async_ld_b128(l + 16, g + 8);
        async_ld_b128(l + 32, g + 16);
        async_ld_b128(l + 48, g + 24);
      } else {
        const float4* src = (const float4*)(w2 + srow + ks + sK);
        #pragma unroll
        for (int c = 0; c < 4; ++c) {
          float4 f0 = src[2*c], f1 = src[2*c + 1];
          v8h h = { (_Float16)f0.x, (_Float16)f0.y, (_Float16)f0.z, (_Float16)f0.w,
                    (_Float16)f1.x, (_Float16)f1.y, (_Float16)f1.z, (_Float16)f1.w };
          *(v8h*)(dst + 8*c) = h;
        }
        if (ks + KSLAB < NH)
          __builtin_prefetch(w2 + srow + ks + KSLAB + sK, 0, 1); // global_prefetch_b8
      }
    };

    stage(0);
    if (useAsync) wait_async0();
    __syncthreads();

    for (int s = 0; s < NSLAB; ++s) {
      const _Float16* cur = Bl + (s & 1) * NTILE * BSTRIDE;
      if (s + 1 < NSLAB) stage(s + 1);           // stream next slab into other buf

      #pragma unroll
      for (int kk = 0; kk < KSLAB; kk += 32) {
        int kg = s * KSLAB + kk;
        v16h b0 = load_b(cur, nWc,      kk, lane);
        v16h b1 = load_b(cur, nWc + 16, kk, lane);
        v16h a0 = load_a(As0, rLoc, kg, lane);
        v16h a1 = load_a(As1, rLoc, kg, lane);
        v16h a2 = load_a(As2, rLoc, kg, lane);
        acc0[0] = wmma_f16(a0, b0, acc0[0]);
        acc0[1] = wmma_f16(a0, b1, acc0[1]);
        acc1[0] = wmma_f16(a1, b0, acc1[0]);
        acc1[1] = wmma_f16(a1, b1, acc1[1]);
        acc2[0] = wmma_f16(a2, b0, acc2[0]);
        acc2[1] = wmma_f16(a2, b1, acc2[1]);
      }

      if (useAsync) wait_async0();               // next slab landed in LDS
      __syncthreads();                           // visible; cur free to reuse
    }

    // ---- fused epilogue: all 8 C elems of a lane share one m ---------------
    int rbase = rLoc + ((lane >> 4) << 3);       // C layout: +8 rows, lanes>=16
    #pragma unroll
    for (int sub = 0; sub < 2; ++sub) {
      int m = mBase + nWc + sub * 16 + (lane & 15);
      float b2m = b2[m], w3m = w3[m];
      #pragma unroll
      for (int r = 0; r < 8; ++r) {
        float z2  = tanhf(acc0[sub][r] + b2m);
        float z2s = z2 * z2;
        float c   = 4.f * w3m * z2 * (1.f - z2s); // 2*w3 * z2 * dz2 * 2 (dh2dz1)
        atomicAdd(&yAcc[rbase + r],  z2s * w3m);
        atomicAdd(&dAcc1[rbase + r], c * acc1[sub][r]);
        atomicAdd(&dAcc2[rbase + r], c * acc2[sub][r]);
      }
    }
  }

  __syncthreads();
  if (t < MTILE) {
    int j = j0 + t;
    out[j]           = yAcc[t] + b3[0];  // y
    out[NXC + j]     = dAcc2[t];         // dydx2  (x2 branch)
    out[2 * NXC + j] = -dAcc1[t];        // -dydx1 (x1 branch, negated)
  }
}

extern "C" void kernel_launch(void* const* d_in, const int* in_sizes, int n_in,
                              void* d_out, int out_size, void* d_ws, size_t ws_size,
                              hipStream_t stream) {
  const float* x    = (const float*)d_in[0];
  const float* w1   = (const float*)d_in[1];
  const float* w1_2 = (const float*)d_in[2];
  const float* b1   = (const float*)d_in[3];
  const float* w2   = (const float*)d_in[4];
  const float* b2   = (const float*)d_in[5];
  const float* w3   = (const float*)d_in[6];
  const float* b3   = (const float*)d_in[7];

  const size_t w2hBytes = (size_t)NH * NH * 2;
  int useAsync = (ws_size >= w2hBytes) ? 1 : 0;
  _Float16* w2h = (_Float16*)d_ws;
  if (useAsync) {
    convert_w2_f16<<<(NH * NH) / (256 * 8), 256, 0, stream>>>(w2, w2h);
  }

  size_t smem = (size_t)(3 * MTILE * ASTRIDE + 2 * NTILE * BSTRIDE) * 2
              + (size_t)(2 * MTILE + 3 * MTILE) * sizeof(float);
  derivnet2d_fused<<<NXC / MTILE, 256, smem, stream>>>(
      x, w1, w1_2, b1, w2, b2, w3, b3, w2h, useAsync, (float*)d_out);
}
